// GSA_21904333210033
// MI455X (gfx1250) — matrix operand order repaired and checked by reading, not compiled, not run on hardware
//
#include <hip/hip_runtime.h>

// ---------------------------------------------------------------------------
// CDNA5 (gfx1250) implementation of the lambda-attention reference.
// All GEMMs use v_wmma_f32_16x16x32_bf16 (bf16 operands, f32 accumulate).
// Softmax / BatchNorm statistics are computed in f32.
// Loads of each k-step are batched ahead of the WMMA group so the scheduler
// can clause them and issue the 4 WMMAs back-to-back.
// ---------------------------------------------------------------------------

typedef __bf16 bf16_t;
typedef __attribute__((ext_vector_type(16))) __bf16 v16bf;
typedef __attribute__((ext_vector_type(8)))  float  v8f;

constexpr int B_   = 8;     // batch
constexpr int C_   = 256;   // input channels
constexpr int S_   = 4096;  // x*y = 64*64
constexpr int NH_  = 64;    // b*heads
constexpr int DK_  = 64;    // dim per head
constexpr int OC3_ = 1536;  // 3*heads*dk
constexpr int HID_ = 512;   // heads*dk
constexpr int OC_  = 256;   // output channels

__device__ __forceinline__ bf16_t f2bf(float f) {
    unsigned u = __builtin_bit_cast(unsigned, f);
    unsigned r = u + 0x7FFFu + ((u >> 16) & 1u);   // round-to-nearest-even
    unsigned short h = (unsigned short)(r >> 16);
    return __builtin_bit_cast(bf16_t, h);
}
__device__ __forceinline__ float bf2f(bf16_t b) {
    unsigned short h = __builtin_bit_cast(unsigned short, b);
    unsigned u = ((unsigned)h) << 16;
    return __builtin_bit_cast(float, u);
}

// A fragment (16x32 bf16, M x K). Lane L: m = L&15, sel = L>>4.
// elements 0..7  -> K = sel*8 + e
// elements 8..15 -> K = 16 + sel*8 + (e-8)
__device__ __forceinline__ v16bf load_frag_a(const bf16_t* __restrict__ base,
                                             int strideM, int strideK) {
    const int lane = threadIdx.x & 31;
    const int m = lane & 15, sel = lane >> 4;
    v16bf f;
#pragma unroll
    for (int e = 0; e < 8; ++e) {
        f[e]     = base[m * strideM + (sel * 8 + e) * strideK];
        f[e + 8] = base[m * strideM + (16 + sel * 8 + e) * strideK];
    }
    return f;
}

// B fragment (32x16 bf16, K x N). Lane L: n = L&15, sel = L>>4.
// elements 0..7  -> K = sel*8 + e
// elements 8..15 -> K = 16 + sel*8 + (e-8)
__device__ __forceinline__ v16bf load_frag_b(const bf16_t* __restrict__ base,
                                             int strideN, int strideK) {
    const int lane = threadIdx.x & 31;
    const int n = lane & 15, sel = lane >> 4;
    v16bf f;
#pragma unroll
    for (int e = 0; e < 8; ++e) {
        f[e]     = base[n * strideN + (sel * 8 + e) * strideK];
        f[e + 8] = base[n * strideN + (16 + sel * 8 + e) * strideK];
    }
    return f;
}

__device__ __forceinline__ v8f wmma_bf(v16bf a, v16bf b, v8f c) {
    return __builtin_amdgcn_wmma_f32_16x16x32_bf16(false, a, false, b,
                                                   (short)0, c, false, false);
}

// Pack 8 f32 accumulator lanes (contiguous output elements) into one b128 store.
__device__ __forceinline__ void store_bf8(bf16_t* dst, const v8f& a) {
    unsigned w0 = ((unsigned)__builtin_bit_cast(unsigned short, f2bf(a[0]))) |
                  ((unsigned)__builtin_bit_cast(unsigned short, f2bf(a[1])) << 16);
    unsigned w1 = ((unsigned)__builtin_bit_cast(unsigned short, f2bf(a[2]))) |
                  ((unsigned)__builtin_bit_cast(unsigned short, f2bf(a[3])) << 16);
    unsigned w2 = ((unsigned)__builtin_bit_cast(unsigned short, f2bf(a[4]))) |
                  ((unsigned)__builtin_bit_cast(unsigned short, f2bf(a[5])) << 16);
    unsigned w3 = ((unsigned)__builtin_bit_cast(unsigned short, f2bf(a[6]))) |
                  ((unsigned)__builtin_bit_cast(unsigned short, f2bf(a[7])) << 16);
    uint4 v = make_uint4(w0, w1, w2, w3);
    *reinterpret_cast<uint4*>(dst) = v;
}

// ------------------------------ prep kernels -------------------------------

__global__ void k_pack_bf(const float* __restrict__ src, bf16_t* __restrict__ dst, int n) {
    int i = blockIdx.x * blockDim.x + threadIdx.x;
    if (i < n) dst[i] = f2bf(src[i]);
}

// P[x][i][d] = rel[i-x][d] if 0 <= i-x < 7 else 0
__global__ void k_build_P(const float* __restrict__ rel, bf16_t* __restrict__ P) {
    int idx = blockIdx.x * blockDim.x + threadIdx.x;
    if (idx >= 64 * 64 * 64) return;
    int d = idx & 63, i = (idx >> 6) & 63, x = idx >> 12;
    int diff = i - x;
    float v = (diff >= 0 && diff < 7) ? rel[diff * 64 + d] : 0.0f;
    P[idx] = f2bf(v);
}

// img [b][c][s] f32 -> img_t [b][s][c] bf16
__global__ void k_img_t(const float* __restrict__ img, bf16_t* __restrict__ img_t) {
    int idx = blockIdx.x * blockDim.x + threadIdx.x;
    if (idx >= B_ * C_ * S_) return;
    int s = idx & (S_ - 1);
    int c = (idx >> 12) & (C_ - 1);
    int b = idx >> 20;
    img_t[(b * S_ + s) * C_ + c] = f2bf(img[idx]);
}

// ------------------------------ qkv GEMM -----------------------------------
// Per batch: (1536 x 256) @ (256 x 4096). Block = 64(M) x 128(N), 8 waves.
// q rows stored transposed: q_t[n][s][d]; k,v rows stored [n][d][s].
__global__ __launch_bounds__(256)
void k_qkv_gemm(const bf16_t* __restrict__ w, const bf16_t* __restrict__ img_t,
                bf16_t* __restrict__ q_t, bf16_t* __restrict__ kb, bf16_t* __restrict__ vb) {
    const int b    = blockIdx.z;
    const int mBlk = blockIdx.y * 64;
    const int nBlk = blockIdx.x * 128;
    const int wv   = threadIdx.x >> 5, lane = threadIdx.x & 31;
    const int mBase = mBlk + (wv & 3) * 16;
    const int nBase = nBlk + (wv >> 2) * 64;
    const bf16_t* imgb = img_t + b * S_ * C_;

    v8f acc[4];
#pragma unroll
    for (int t = 0; t < 4; ++t)
#pragma unroll
        for (int r = 0; r < 8; ++r) acc[t][r] = 0.f;

#pragma unroll 2
    for (int k0 = 0; k0 < C_; k0 += 32) {
        v16bf a  = load_frag_a(w + mBase * C_ + k0, C_, 1);
        v16bf b0 = load_frag_b(imgb + (nBase +  0) * C_ + k0, C_, 1);
        v16bf b1 = load_frag_b(imgb + (nBase + 16) * C_ + k0, C_, 1);
        v16bf b2 = load_frag_b(imgb + (nBase + 32) * C_ + k0, C_, 1);
        v16bf b3 = load_frag_b(imgb + (nBase + 48) * C_ + k0, C_, 1);
        acc[0] = wmma_bf(a, b0, acc[0]);
        acc[1] = wmma_bf(a, b1, acc[1]);
        acc[2] = wmma_bf(a, b2, acc[2]);
        acc[3] = wmma_bf(a, b3, acc[3]);
    }

    const int sec   = mBase / 512;          // 0=q, 1=k, 2=v
    const int hh    = (mBase % 512) / 64;
    const int nq    = b * 8 + hh;
    const int dBase = mBase % 64;
    const int mnL = lane & 15, selL = lane >> 4;
#pragma unroll
    for (int t = 0; t < 4; ++t) {
        const int sB = nBase + t * 16 + mnL;
        if (sec == 0) {
            // 8 contiguous bf16 -> single b128 store (16B aligned).
            store_bf8(q_t + (nq * S_ + sB) * DK_ + dBase + selL * 8, acc[t]);
        } else {
            bf16_t* dst = (sec == 1) ? kb : vb;
            int base = (nq * DK_ + dBase + selL * 8) * S_ + sB;
#pragma unroll
            for (int r = 0; r < 8; ++r) dst[base + r * S_] = f2bf(acc[t][r]);
        }
    }
}

// --------------------------- softmax over y --------------------------------
// k[n][d][x][y]: softmax over last 64, in place. One row per wave.
__global__ void k_softmax(bf16_t* __restrict__ kb) {
    const int wv = threadIdx.x >> 5, lane = threadIdx.x & 31;
    const int row = blockIdx.x * 8 + wv;     // 262144 rows
    bf16_t* p = kb + row * 64;
    float v0 = bf2f(p[lane]), v1 = bf2f(p[lane + 32]);
    float mx = fmaxf(v0, v1);
#pragma unroll
    for (int m = 16; m >= 1; m >>= 1) mx = fmaxf(mx, __shfl_xor(mx, m, 32));
    float e0 = __expf(v0 - mx), e1 = __expf(v1 - mx);
    float sm = e0 + e1;
#pragma unroll
    for (int m = 16; m >= 1; m >>= 1) sm += __shfl_xor(sm, m, 32);
    float inv = 1.0f / sm;
    p[lane]      = f2bf(e0 * inv);
    p[lane + 32] = f2bf(e1 * inv);
}

// ----------------------------- context -------------------------------------
// ctxT[n][e][d] = sum_s v[n][e][s] * ksm[n][d][s]     (K = 4096)
__global__ __launch_bounds__(256)
void k_context(const bf16_t* __restrict__ ksm, const bf16_t* __restrict__ vb,
               bf16_t* __restrict__ ctxT) {
    const int n = blockIdx.x;
    const int wv = threadIdx.x >> 5, lane = threadIdx.x & 31;
    const int tm = wv & 3, tn0 = (wv >> 2) * 2;
    const bf16_t* va = vb  + n * DK_ * S_ + (tm * 16) * S_;
    const bf16_t* ka = ksm + n * DK_ * S_;

    v8f acc[2];
#pragma unroll
    for (int t = 0; t < 2; ++t)
#pragma unroll
        for (int r = 0; r < 8; ++r) acc[t][r] = 0.f;

#pragma unroll 2
    for (int k0 = 0; k0 < S_; k0 += 32) {
        // prefetch next k-step lines into cache (global_prefetch_b8)
        if (k0 + 64 < S_) {
            __builtin_prefetch(va + (lane & 15) * S_ + k0 + 64, 0, 1);
            __builtin_prefetch(ka + ((tn0 * 16) + (lane & 15)) * S_ + k0 + 64, 0, 1);
        }
        v16bf a  = load_frag_a(va + k0, S_, 1);
        v16bf b0 = load_frag_b(ka + ((tn0 + 0) * 16) * S_ + k0, S_, 1);
        v16bf b1 = load_frag_b(ka + ((tn0 + 1) * 16) * S_ + k0, S_, 1);
        acc[0] = wmma_bf(a, b0, acc[0]);
        acc[1] = wmma_bf(a, b1, acc[1]);
    }
    const int mnL = lane & 15, selL = lane >> 4;
#pragma unroll
    for (int t = 0; t < 2; ++t)
#pragma unroll
        for (int r = 0; r < 8; ++r) {
            int e = tm * 16 + selL * 8 + r, d = (tn0 + t) * 16 + mnL;
            ctxT[(n * DK_ + e) * DK_ + d] = f2bf(acc[t][r]);
        }
}

// ----------------------------- content -------------------------------------
// cont[n][e][s] = sum_d ctxT[n][e][d] * q_t[n][s][d]   (K = 64)
__global__ __launch_bounds__(256)
void k_content(const bf16_t* __restrict__ ctxT, const bf16_t* __restrict__ q_t,
               float* __restrict__ cont) {
    const int n = blockIdx.y;
    const int nBlk = blockIdx.x * 128;
    const int wv = threadIdx.x >> 5, lane = threadIdx.x & 31;
    const int mBase = (wv & 3) * 16;
    const int nBase = nBlk + (wv >> 2) * 64;
    const bf16_t* ca = ctxT + n * DK_ * DK_;
    const bf16_t* qb = q_t + n * S_ * DK_;

    v8f acc[4];
#pragma unroll
    for (int t = 0; t < 4; ++t)
#pragma unroll
        for (int r = 0; r < 8; ++r) acc[t][r] = 0.f;

#pragma unroll
    for (int k0 = 0; k0 < DK_; k0 += 32) {
        v16bf a  = load_frag_a(ca + mBase * DK_ + k0, DK_, 1);
        v16bf b0 = load_frag_b(qb + (nBase +  0) * DK_ + k0, DK_, 1);
        v16bf b1 = load_frag_b(qb + (nBase + 16) * DK_ + k0, DK_, 1);
        v16bf b2 = load_frag_b(qb + (nBase + 32) * DK_ + k0, DK_, 1);
        v16bf b3 = load_frag_b(qb + (nBase + 48) * DK_ + k0, DK_, 1);
        acc[0] = wmma_bf(a, b0, acc[0]);
        acc[1] = wmma_bf(a, b1, acc[1]);
        acc[2] = wmma_bf(a, b2, acc[2]);
        acc[3] = wmma_bf(a, b3, acc[3]);
    }
    const int mnL = lane & 15, selL = lane >> 4;
#pragma unroll
    for (int t = 0; t < 4; ++t)
#pragma unroll
        for (int r = 0; r < 8; ++r) {
            int e = mBase + selL * 8 + r, s = nBase + t * 16 + mnL;
            cont[(n * DK_ + e) * S_ + s] = acc[t][r];
        }
}

// ------------------------------- S = P @ q ---------------------------------
// Sout[n][i][x*64+y] = sum_d P[x][i][d] * q_t[n][x*64+y][d]   (K = 64)
__global__ __launch_bounds__(256)
void k_sxy(const bf16_t* __restrict__ P, const bf16_t* __restrict__ q_t,
           bf16_t* __restrict__ Sout) {
    const int x = blockIdx.x, n = blockIdx.y;
    const int wv = threadIdx.x >> 5, lane = threadIdx.x & 31;
    const int tm = wv & 3, tn0 = (wv >> 2) * 2;
    const bf16_t* Pa = P + x * DK_ * DK_;
    const bf16_t* qb = q_t + (n * S_ + x * 64) * DK_;

    v8f acc[2];
#pragma unroll
    for (int t = 0; t < 2; ++t)
#pragma unroll
        for (int r = 0; r < 8; ++r) acc[t][r] = 0.f;

#pragma unroll
    for (int k0 = 0; k0 < DK_; k0 += 32) {
        v16bf a  = load_frag_a(Pa + (tm * 16) * DK_ + k0, DK_, 1);
        v16bf b0 = load_frag_b(qb + ((tn0 + 0) * 16) * DK_ + k0, DK_, 1);
        v16bf b1 = load_frag_b(qb + ((tn0 + 1) * 16) * DK_ + k0, DK_, 1);
        acc[0] = wmma_bf(a, b0, acc[0]);
        acc[1] = wmma_bf(a, b1, acc[1]);
    }
    const int mnL = lane & 15, selL = lane >> 4;
#pragma unroll
    for (int t = 0; t < 2; ++t)
#pragma unroll
        for (int r = 0; r < 8; ++r) {
            int i = tm * 16 + selL * 8 + r, y = (tn0 + t) * 16 + mnL;
            Sout[(n * DK_ + i) * S_ + x * 64 + y] = f2bf(acc[t][r]);
        }
}

// ------------------------------ Yh GEMM ------------------------------------
// Yh[n][e][x*64+y] = sum_i v[n][e][i*64+y] * Sx[n][i][x*64+y]   (K = 64)
__global__ __launch_bounds__(256)
void k_yh(const bf16_t* __restrict__ vb, const bf16_t* __restrict__ Sx,
          float* __restrict__ Yh) {
    const int y = blockIdx.x, n = blockIdx.y;
    const int wv = threadIdx.x >> 5, lane = threadIdx.x & 31;
    const int tm = wv & 3, tn0 = (wv >> 2) * 2;
    const bf16_t* va = vb + n * DK_ * S_ + y;   // A: (m=e, k=i): [m*S_ + k*64]
    const bf16_t* sb = Sx + n * DK_ * S_ + y;   // B: (k=i, nn=x): [nn*64 + k*S_]

    v8f acc[2];
#pragma unroll
    for (int t = 0; t < 2; ++t)
#pragma unroll
        for (int r = 0; r < 8; ++r) acc[t][r] = 0.f;

#pragma unroll
    for (int k0 = 0; k0 < DK_; k0 += 32) {
        v16bf a  = load_frag_a(va + (tm * 16) * S_ + k0 * 64, S_, 64);
        v16bf b0 = load_frag_b(sb + ((tn0 + 0) * 16) * 64 + k0 * S_, 64, S_);
        v16bf b1 = load_frag_b(sb + ((tn0 + 1) * 16) * 64 + k0 * S_, 64, S_);
        acc[0] = wmma_bf(a, b0, acc[0]);
        acc[1] = wmma_bf(a, b1, acc[1]);
    }
    const int mnL = lane & 15, selL = lane >> 4;
#pragma unroll
    for (int t = 0; t < 2; ++t)
#pragma unroll
        for (int r = 0; r < 8; ++r) {
            int e = tm * 16 + selL * 8 + r, xx = (tn0 + t) * 16 + mnL;
            Yh[(n * DK_ + e) * S_ + xx * 64 + y] = acc[t][r];
        }
}

// ----------------------------- BatchNorm -----------------------------------
__global__ void k_bn_stats(const float* __restrict__ Yh, float* __restrict__ stats) {
    const int e = blockIdx.x;   // 64 channels
    float s = 0.f, sq = 0.f;
    for (int n = 0; n < NH_; ++n) {
        const float* p = Yh + (n * DK_ + e) * S_;
        for (int i = threadIdx.x; i < S_; i += blockDim.x) {
            float v = p[i];
            s += v; sq += v * v;
        }
    }
    __shared__ float sh[256], sh2[256];
    sh[threadIdx.x] = s; sh2[threadIdx.x] = sq;
    __syncthreads();
    for (int o = 128; o > 0; o >>= 1) {
        if ((int)threadIdx.x < o) {
            sh[threadIdx.x]  += sh[threadIdx.x + o];
            sh2[threadIdx.x] += sh2[threadIdx.x + o];
        }
        __syncthreads();
    }
    if (threadIdx.x == 0) {
        const float cnt = (float)NH_ * (float)S_;
        float mean = sh[0] / cnt;
        float var  = sh2[0] / cnt - mean * mean;
        stats[e * 2]     = mean;
        stats[e * 2 + 1] = rsqrtf(var + 1e-5f);
    }
}

__global__ void k_bn_apply(const float* __restrict__ Yh, const float* __restrict__ stats,
                           const float* __restrict__ gamma, const float* __restrict__ beta,
                           bf16_t* __restrict__ Yhbn) {
    int idx = blockIdx.x * blockDim.x + threadIdx.x;
    if (idx >= NH_ * DK_ * S_) return;
    int e = (idx >> 12) & 63;
    float v = (Yh[idx] - stats[e * 2]) * stats[e * 2 + 1] * gamma[e] + beta[e];
    Yhbn[idx] = f2bf(v);
}

// --------------------------- rel path (fused add) --------------------------
// cont[n][e][x*64+y] += sum_i Yhbn[n][e][i*64+y] * Sy[n][i][x*64+y]
// C accumulator initialized from cont, D written back to cont.
__global__ __launch_bounds__(256)
void k_rel(const bf16_t* __restrict__ Yhbn, const bf16_t* __restrict__ Sy,
           float* __restrict__ cont) {
    const int y = blockIdx.x, n = blockIdx.y;
    const int wv = threadIdx.x >> 5, lane = threadIdx.x & 31;
    const int tm = wv & 3, tn0 = (wv >> 2) * 2;
    const bf16_t* va = Yhbn + n * DK_ * S_ + y;
    const bf16_t* sb = Sy   + n * DK_ * S_ + y;
    float* cb = cont + n * DK_ * S_ + y;
    const int mnL = lane & 15, selL = lane >> 4;

    v8f acc[2];
#pragma unroll
    for (int t = 0; t < 2; ++t)
#pragma unroll
        for (int r = 0; r < 8; ++r) {
            int e = tm * 16 + selL * 8 + r, xx = (tn0 + t) * 16 + mnL;
            acc[t][r] = cb[e * S_ + xx * 64];
        }

#pragma unroll
    for (int k0 = 0; k0 < DK_; k0 += 32) {
        v16bf a  = load_frag_a(va + (tm * 16) * S_ + k0 * 64, S_, 64);
        v16bf b0 = load_frag_b(sb + ((tn0 + 0) * 16) * 64 + k0 * S_, 64, S_);
        v16bf b1 = load_frag_b(sb + ((tn0 + 1) * 16) * 64 + k0 * S_, 64, S_);
        acc[0] = wmma_bf(a, b0, acc[0]);
        acc[1] = wmma_bf(a, b1, acc[1]);
    }
#pragma unroll
    for (int t = 0; t < 2; ++t)
#pragma unroll
        for (int r = 0; r < 8; ++r) {
            int e = tm * 16 + selL * 8 + r, xx = (tn0 + t) * 16 + mnL;
            cb[e * S_ + xx * 64] = acc[t][r];
        }
}

// cont[n][e][s] f32 -> cont_t[b][s][h*64+e] bf16
__global__ void k_pack_cont(const float* __restrict__ cont, bf16_t* __restrict__ cont_t) {
    int idx = blockIdx.x * blockDim.x + threadIdx.x;
    if (idx >= NH_ * DK_ * S_) return;
    int s = idx & (S_ - 1);
    int e = (idx >> 12) & 63;
    int n = idx >> 18;
    int b = n >> 3, h = n & 7;
    cont_t[(b * S_ + s) * HID_ + h * 64 + e] = f2bf(cont[idx]);
}

// --------------------------- output projection -----------------------------
// out[b][o][s] = sum_hc w_out[o][hc] * cont_t[b][s][hc] + bias[o]   (K = 512)
__global__ __launch_bounds__(256)
void k_out_gemm(const bf16_t* __restrict__ wo, const bf16_t* __restrict__ cont_t,
                const float* __restrict__ bias, float* __restrict__ out) {
    const int b = blockIdx.z;
    const int mBlk = blockIdx.y * 64;
    const int nBlk = blockIdx.x * 128;
    const int wv = threadIdx.x >> 5, lane = threadIdx.x & 31;
    const int mBase = mBlk + (wv & 3) * 16;
    const int nBase = nBlk + (wv >> 2) * 64;
    const bf16_t* cb = cont_t + b * S_ * HID_;

    v8f acc[4];
#pragma unroll
    for (int t = 0; t < 4; ++t)
#pragma unroll
        for (int r = 0; r < 8; ++r) acc[t][r] = 0.f;

#pragma unroll 2
    for (int k0 = 0; k0 < HID_; k0 += 32) {
        v16bf a  = load_frag_a(wo + mBase * HID_ + k0, HID_, 1);
        v16bf b0 = load_frag_b(cb + (nBase +  0) * HID_ + k0, HID_, 1);
        v16bf b1 = load_frag_b(cb + (nBase + 16) * HID_ + k0, HID_, 1);
        v16bf b2 = load_frag_b(cb + (nBase + 32) * HID_ + k0, HID_, 1);
        v16bf b3 = load_frag_b(cb + (nBase + 48) * HID_ + k0, HID_, 1);
        acc[0] = wmma_bf(a, b0, acc[0]);
        acc[1] = wmma_bf(a, b1, acc[1]);
        acc[2] = wmma_bf(a, b2, acc[2]);
        acc[3] = wmma_bf(a, b3, acc[3]);
    }
    const int mnL = lane & 15, selL = lane >> 4;
#pragma unroll
    for (int t = 0; t < 4; ++t)
#pragma unroll
        for (int r = 0; r < 8; ++r) {
            int o = mBase + selL * 8 + r, s = nBase + t * 16 + mnL;
            out[(b * OC_ + o) * S_ + s] = acc[t][r] + bias[o];
        }
}

// ---------------------------------------------------------------------------

extern "C" void kernel_launch(void* const* d_in, const int* in_sizes, int n_in,
                              void* d_out, int out_size, void* d_ws, size_t ws_size,
                              hipStream_t stream) {
    (void)in_sizes; (void)n_in; (void)out_size; (void)ws_size;
    const float* img   = (const float*)d_in[0];
    const float* w_qkv = (const float*)d_in[1];
    const float* w_out = (const float*)d_in[2];
    const float* b_out = (const float*)d_in[3];
    const float* gamma = (const float*)d_in[4];
    const float* beta  = (const float*)d_in[5];
    const float* rel_r = (const float*)d_in[6];
    const float* rel_c = (const float*)d_in[7];
    float* out = (float*)d_out;

    char* ws = (char*)d_ws;
    size_t off = 0;
    auto take = [&](size_t bytes) {
        char* p = ws + off;
        off += (bytes + 255) & ~(size_t)255;
        return p;
    };
    bf16_t* imgT  = (bf16_t*)take((size_t)B_ * S_ * C_ * 2);      // 16.8 MB
    bf16_t* wqkvB = (bf16_t*)take((size_t)OC3_ * C_ * 2);         //  0.8 MB
    bf16_t* woutB = (bf16_t*)take((size_t)OC_ * HID_ * 2);        //  0.3 MB
    bf16_t* Px    = (bf16_t*)take((size_t)64 * 64 * 64 * 2);      //  0.5 MB
    bf16_t* Py    = (bf16_t*)take((size_t)64 * 64 * 64 * 2);      //  0.5 MB
    bf16_t* qT    = (bf16_t*)take((size_t)NH_ * S_ * DK_ * 2);    // 33.6 MB (later cont_t)
    bf16_t* kB    = (bf16_t*)take((size_t)NH_ * DK_ * S_ * 2);    // 33.6 MB (later Sx, Sy)
    bf16_t* vB    = (bf16_t*)take((size_t)NH_ * DK_ * S_ * 2);    // 33.6 MB (later Yh_bn)
    bf16_t* ctxT  = (bf16_t*)take((size_t)NH_ * DK_ * DK_ * 2);   //  0.5 MB
    float*  cont  = (float*)take((size_t)NH_ * DK_ * S_ * 4);     // 67.1 MB
    float*  Yh    = (float*)take((size_t)NH_ * DK_ * S_ * 4);     // 67.1 MB
    float*  stats = (float*)take((size_t)DK_ * 2 * 4);
    // buffer reuse across the (sequential) pipeline:
    bf16_t* Sx    = kB;   // k dead after context GEMM
    bf16_t* Sy    = kB;   // Sx dead after Yh GEMM
    bf16_t* Yhbn  = vB;   // v dead after Yh GEMM
    bf16_t* contT = qT;   // q dead after second S GEMM

    const dim3 blk(256);

    // --- preparation
    k_pack_bf<<<dim3((OC3_ * C_ + 255) / 256), blk, 0, stream>>>(w_qkv, wqkvB, OC3_ * C_);
    k_pack_bf<<<dim3((OC_ * HID_ + 255) / 256), blk, 0, stream>>>(w_out, woutB, OC_ * HID_);
    k_build_P<<<dim3(1024), blk, 0, stream>>>(rel_r, Px);
    k_build_P<<<dim3(1024), blk, 0, stream>>>(rel_c, Py);
    k_img_t<<<dim3(B_ * C_ * S_ / 256), blk, 0, stream>>>(img, imgT);

    // --- qkv projection (WMMA), split into q_t / k / v
    k_qkv_gemm<<<dim3(32, 24, 8), blk, 0, stream>>>(wqkvB, imgT, qT, kB, vB);

    // --- softmax over y (in place on k)
    k_softmax<<<dim3(NH_ * DK_ * 64 / 8), blk, 0, stream>>>(kB);

    // --- context = ksm . v  (per n, K=4096)
    k_context<<<dim3(NH_), blk, 0, stream>>>(kB, vB, ctxT);

    // --- content_out = ctx^T . q
    k_content<<<dim3(32, NH_), blk, 0, stream>>>(ctxT, qT, cont);

    // --- Sx = Px . q ; Yh = v . Sx
    k_sxy<<<dim3(64, NH_), blk, 0, stream>>>(Px, qT, Sx);
    k_yh<<<dim3(64, NH_), blk, 0, stream>>>(vB, Sx, Yh);

    // --- BatchNorm over (n, x, y) per channel
    k_bn_stats<<<dim3(DK_), blk, 0, stream>>>(Yh, stats);
    k_bn_apply<<<dim3(NH_ * DK_ * S_ / 256), blk, 0, stream>>>(Yh, stats, gamma, beta, Yhbn);

    // --- Sy = Py . q ; cont += Yh_bn . Sy (fused via WMMA C accumulator)
    k_sxy<<<dim3(64, NH_), blk, 0, stream>>>(Py, qT, Sy);
    k_rel<<<dim3(64, NH_), blk, 0, stream>>>(Yhbn, Sy, cont);

    // --- pack to [b][s][512] and project to output channels (+bias)
    k_pack_cont<<<dim3(NH_ * DK_ * S_ / 256), blk, 0, stream>>>(cont, contT);
    k_out_gemm<<<dim3(32, 4, 8), blk, 0, stream>>>(woutB, contT, b_out, out);
}
